// AttentionGuidedNet_35278861369925
// MI455X (gfx1250) — matrix operand order repaired and checked by reading, not compiled, not run on hardware
//
#include <hip/hip_runtime.h>
#include <hip/hip_bf16.h>
#include <math.h>

// ---------------------------------------------------------------------------
// AttentionGuidedNet forward on MI455X (gfx1250, wave32).
// Convolutions = implicit GEMM on v_wmma_f32_16x16x32_f16, fully-static shape
// templates, M64xN32 block tile (2 WMMAs/wave/K-step sharing one A fragment),
// LDS im2col tile stored in B-fragment order (1x 32B read per fragment).
// ---------------------------------------------------------------------------

typedef _Float16 v16h __attribute__((ext_vector_type(16)));
typedef float    v8f  __attribute__((ext_vector_type(8)));

#define BATCH 4
#define NCLS  14
#define HW    224
#define PIX   (HW * HW)          // 50176
#define NLAB  (PIX + 1)          // 50177

// ============================ conv as WMMA GEMM ============================
// Y[b,m,n] = relu( sum_k W[m,k] * im2col(X)[k,n] ), m=Cout, n=OH*OW.
// Block = 128 threads = 4 waves. Block tile: M=64 (16 rows/wave), N=32.
// LDS holds the 32(K)x32(N) f16 im2col tile pre-swizzled into WMMA B-fragment
// order: element (k,n) lives at halfword ((n>>4)*32 + (k>>4)*16 + (n&15))*16
// + (k&15), so lane L of fragment-group g reads 16 contiguous halfs (32B).
template <int Cin, int IH, int IW, int Cout, int OH, int OW,
          int KH, int KW, int STR, int PAD>
__global__ __launch_bounds__(128)
void conv_wmma_kernel(const float* __restrict__ X, const float* __restrict__ Wt,
                      float* __restrict__ Y)
{
    constexpr int KHW = KH * KW;
    constexpr int K   = Cin * KHW;
    constexpr int N   = OH * OW;

    const int batch = blockIdx.z;
    const int n0    = blockIdx.x * 32;
    const int wave  = threadIdx.x >> 5;
    const int lane  = threadIdx.x & 31;
    const int m0    = (blockIdx.y * 4 + wave) * 16;   // Cout multiple of 64
    const int tid   = threadIdx.x;
    const int half  = lane >> 4;
    const int nlane = lane & 15;
    const float* Xb   = X + (size_t)batch * Cin * IH * IW;
    const float* Arow = Wt + (size_t)(m0 + nlane) * K;

    __shared__ alignas(32) _Float16 Bs[2 * 32 * 16];  // fragment-order B tile

    // Per-thread gather geometry: col is fixed (tid&31); k slots are
    // (tid>>5) + 4*i, i=0..7 -> 128 threads x 8 slots cover the 32x32 tile.
    const int  nf   = tid & 31;
    const int  nn_f = n0 + nf;
    const bool nok  = nn_f < N;
    const int  oh_f = nn_f / OW;                      // constant divisor
    const int  ow_f = nn_f - oh_f * OW;
    int kslot[8], laddr[8];
    #pragma unroll
    for (int i = 0; i < 8; ++i) {
        int k    = (tid >> 5) + 4 * i;                // 0..31
        kslot[i] = k;
        laddr[i] = (((nf >> 4) * 32 + (k >> 4) * 16 + (nf & 15)) << 4) + (k & 15);
    }

    v8f acc0 = {};
    v8f acc1 = {};
    for (int kb = 0; kb < K; kb += 32) {
        __syncthreads();                              // WAR vs previous reads
        // cooperative im2col gather into fragment-order LDS tile
        #pragma unroll
        for (int i = 0; i < 8; ++i) {
            int kk = kb + kslot[i];
            float val = 0.0f;
            if ((K % 32 == 0 || kk < K) && nok) {
                int ci  = kk / KHW;                   // constant divisor
                int rem = kk - ci * KHW;
                int kh  = rem / KW;
                int kw  = rem - kh * KW;
                int ih  = oh_f * STR - PAD + kh;
                int iw  = ow_f * STR - PAD + kw;
                if (ih >= 0 && ih < IH && iw >= 0 && iw < IW)
                    val = Xb[((size_t)ci * IH + ih) * IW + iw];
            }
            Bs[laddr[i]] = (_Float16)val;
        }
        __syncthreads();

        // A fragment: element pairs (2t,2t+1) are k-contiguous -> float2 loads
        v16h afrag;
        #pragma unroll
        for (int t = 0; t < 8; ++t) {
            int ka  = 16 * half + 2 * t;
            int kkA = kb + ka;
            float w0, w1;
            if constexpr (K % 32 == 0) {
                float2 w2 = *(const float2*)(Arow + kkA);
                w0 = w2.x; w1 = w2.y;
            } else {
                w0 = (kkA     < K) ? Arow[kkA]     : 0.0f;
                w1 = (kkA + 1 < K) ? Arow[kkA + 1] : 0.0f;
            }
            afrag[2 * t]     = (_Float16)w0;
            afrag[2 * t + 1] = (_Float16)w1;
        }
        __builtin_prefetch(Arow + kb + 64, 0, 1);     // A chunk two K-steps ahead -> L2

        // B fragments: one contiguous 32B LDS read per fragment
        const v16h* Bv = (const v16h*)Bs;
        v16h b0 = Bv[lane];
        v16h b1 = Bv[32 + lane];

        acc0 = __builtin_amdgcn_wmma_f32_16x16x32_f16(
                   false, afrag, false, b0, (short)0, acc0, false, false);
        acc1 = __builtin_amdgcn_wmma_f32_16x16x32_f16(
                   false, afrag, false, b1, (short)0, acc1, false, false);
    }

    // store with fused ReLU; D layout: VGPR r -> row m0 + r + 8*half, col nlane
    float* Yb = Y + (size_t)batch * Cout * N;
    const int nn0 = n0 + nlane;
    const int nn1 = n0 + 16 + nlane;
    #pragma unroll
    for (int r = 0; r < 8; ++r) {
        int m = m0 + r + 8 * half;
        if (nn0 < N) {
            float v = acc0[r];
            Yb[(size_t)m * N + nn0] = v > 0.0f ? v : 0.0f;
        }
        if (nn1 < N) {
            float v = acc1[r];
            Yb[(size_t)m * N + nn1] = v > 0.0f ? v : 0.0f;
        }
    }
}

// ============================ small kernels ================================
__global__ void maxpool49_kernel(const float* __restrict__ feat,
                                 float* __restrict__ pool, int total)
{
    int i = blockIdx.x * blockDim.x + threadIdx.x;
    if (i >= total) return;
    const float* p = feat + (size_t)i * 49;
    float m = p[0];
    for (int k = 1; k < 49; ++k) m = fmaxf(m, p[k]);
    pool[i] = m;
}

// out[b,c] = sigmoid(bias[c] + pa[b]·W[c,:na] + pb[b]·W[c,na:na+nb])
__global__ void fc_sigmoid_kernel(const float* __restrict__ pa,
                                  const float* __restrict__ pb,
                                  int na, int nb,
                                  const float* __restrict__ W,
                                  const float* __restrict__ bias,
                                  float* __restrict__ out, int BC, int C)
{
    int i = blockIdx.x * blockDim.x + threadIdx.x;
    if (i >= BC) return;
    int b = i / C, c = i % C;
    const float* w = W + (size_t)c * (na + nb);
    float acc = bias[c];
    for (int k = 0; k < na; ++k) acc += pa[b * na + k] * w[k];
    for (int k = 0; k < nb; ++k) acc += pb[b * nb + k] * w[na + k];
    out[i] = 1.0f / (1.0f + expf(-acc));
}

__global__ void hm_max_kernel(const float* __restrict__ feat,
                              float* __restrict__ hm7, int total)
{
    int i = blockIdx.x * blockDim.x + threadIdx.x;   // b*49 + pos
    if (i >= total) return;
    int b = i / 49, pos = i % 49;
    const float* f = feat + ((size_t)b * 2048) * 49 + pos;
    float m = 0.0f;
    for (int c = 0; c < 2048; ++c) m = fmaxf(m, fabsf(f[(size_t)c * 49]));
    hm7[i] = m;
}

__global__ void hm_norm_kernel(float* hm7)
{
    int b = threadIdx.x;
    if (b >= BATCH) return;
    float* h = hm7 + b * 49;
    float mn = h[0], mx = h[0];
    for (int i = 1; i < 49; ++i) { mn = fminf(mn, h[i]); mx = fmaxf(mx, h[i]); }
    float inv = 1.0f / (mx - mn);
    for (int i = 0; i < 49; ++i) h[i] = (h[i] - mn) * inv;
}

// bilinear 7x7 -> 224x224, threshold, seed labels (row-major id+1)
__global__ void resize_mask_kernel(const float* __restrict__ hm7,
                                   int* __restrict__ mask,
                                   int* __restrict__ lab, int total)
{
    int i = blockIdx.x * blockDim.x + threadIdx.x;
    if (i >= total) return;
    int b = i / PIX, r = i % PIX, y = r / HW, x = r % HW;
    float sy = (y + 0.5f) * (7.0f / 224.0f) - 0.5f;
    float sx = (x + 0.5f) * (7.0f / 224.0f) - 0.5f;
    sy = fminf(fmaxf(sy, 0.0f), 6.0f);
    sx = fminf(fmaxf(sx, 0.0f), 6.0f);
    int y0 = (int)floorf(sy), x0 = (int)floorf(sx);
    int y1 = min(y0 + 1, 6),  x1 = min(x0 + 1, 6);
    float fy = sy - (float)y0, fx = sx - (float)x0;
    const float* h = hm7 + b * 49;
    float v = (1.0f - fy) * ((1.0f - fx) * h[y0 * 7 + x0] + fx * h[y0 * 7 + x1])
            +          fy * ((1.0f - fx) * h[y1 * 7 + x0] + fx * h[y1 * 7 + x1]);
    int m = (v > 0.7f) ? 1 : 0;
    mask[i] = m;
    lab[i]  = m * (r + 1);
}

// one step of masked 3x3 max-label propagation (zero-padded SAME window)
__global__ void dilate_kernel(const int* __restrict__ src,
                              int* __restrict__ dst,
                              const int* __restrict__ mask, int total)
{
    int i = blockIdx.x * blockDim.x + threadIdx.x;
    if (i >= total) return;
    int b = i / PIX, r = i % PIX, y = r / HW, x = r % HW;
    const int* S = src + b * PIX;
    int best = 0;
    #pragma unroll
    for (int dy = -1; dy <= 1; ++dy) {
        int yy = y + dy;
        if (yy < 0 || yy >= HW) continue;
        #pragma unroll
        for (int dx = -1; dx <= 1; ++dx) {
            int xx = x + dx;
            if (xx < 0 || xx >= HW) continue;
            best = max(best, S[yy * HW + xx]);
        }
    }
    dst[i] = mask[i] ? best : 0;
}

__global__ void zero_counts_kernel(int* __restrict__ counts, int total)
{
    int i = blockIdx.x * blockDim.x + threadIdx.x;
    if (i >= total) return;
    counts[i] = (i % NLAB == 0) ? -1 : 0;   // background slot forced to -1
}

__global__ void count_labels_kernel(const int* __restrict__ lab,
                                    int* __restrict__ counts, int total)
{
    int i = blockIdx.x * blockDim.x + threadIdx.x;
    if (i >= total) return;
    int l = lab[i];
    if (l > 0) atomicAdd(&counts[(i / PIX) * NLAB + l], 1);
}

__global__ void argmax_kernel(const int* __restrict__ counts,
                              int* __restrict__ maxlab, int* __restrict__ bbox)
{
    int b = threadIdx.x;
    if (b >= BATCH) return;
    const int* c = counts + b * NLAB;
    int best = c[0], bi = 0;
    for (int i = 1; i < NLAB; ++i) { int v = c[i]; if (v > best) { best = v; bi = i; } }
    maxlab[b] = bi;
    bbox[b * 4 + 0] = HW;  bbox[b * 4 + 1] = -1;   // xmin / xmax (rows)
    bbox[b * 4 + 2] = HW;  bbox[b * 4 + 3] = -1;   // ymin / ymax (cols)
}

__global__ void bbox_kernel(const int* __restrict__ lab,
                            const int* __restrict__ maxlab,
                            int* __restrict__ bbox)
{
    int b = blockIdx.x, i = threadIdx.x;           // 224 threads
    const int* L = lab + b * PIX;
    int ml = maxlab[b];
    bool rowAny = false, colAny = false;
    for (int j = 0; j < HW; ++j) {
        if (L[i * HW + j] == ml) rowAny = true;
        if (L[j * HW + i] == ml) colAny = true;
    }
    if (rowAny) { atomicMin(&bbox[b * 4 + 0], i); atomicMax(&bbox[b * 4 + 1], i); }
    if (colAny) { atomicMin(&bbox[b * 4 + 2], i); atomicMax(&bbox[b * 4 + 3], i); }
}

__global__ void crop_resize_kernel(const float* __restrict__ img,
                                   const int* __restrict__ bbox,
                                   float* __restrict__ patch, int total)
{
    int i = blockIdx.x * blockDim.x + threadIdx.x;
    if (i >= total) return;
    int b  = i / (3 * PIX);
    int rm = i % (3 * PIX);
    int c  = rm / PIX;
    int r2 = rm % PIX;
    int y  = r2 / HW, x = r2 % HW;
    int xmin = bbox[b * 4 + 0], xmax = bbox[b * 4 + 1];
    int ymin = bbox[b * 4 + 2], ymax = bbox[b * 4 + 3];
    float ch = fmaxf((float)(xmax - xmin), 1.0f);
    float cw = fmaxf((float)(ymax - ymin), 1.0f);
    float fy0 = (float)xmin, fx0 = (float)ymin;
    float oy = (y + 0.5f) * ch * (1.0f / 224.0f) + fy0 - 0.5f;
    float ox = (x + 0.5f) * cw * (1.0f / 224.0f) + fx0 - 0.5f;
    oy = fminf(fmaxf(oy, fy0), fmaxf(fy0 + ch - 1.0f, fy0));
    ox = fminf(fmaxf(ox, fx0), fmaxf(fx0 + cw - 1.0f, fx0));
    int y0 = (int)floorf(oy); float fy = oy - (float)y0;
    int x0 = (int)floorf(ox); float fx = ox - (float)x0;
    int y0c = min(max(y0, 0), HW - 1), y1c = min(max(y0 + 1, 0), HW - 1);
    int x0c = min(max(x0, 0), HW - 1), x1c = min(max(x0 + 1, 0), HW - 1);
    const float* im = img + ((size_t)b * 3 + c) * PIX;
    float v = (1.0f - fy) * ((1.0f - fx) * im[y0c * HW + x0c] + fx * im[y0c * HW + x1c])
            +          fy * ((1.0f - fx) * im[y1c * HW + x0c] + fx * im[y1c * HW + x1c]);
    patch[i] = v;
}

__device__ __forceinline__ float bce_term(float p, float t)
{
    p = fminf(fmaxf(p, 1e-7f), 1.0f - 1e-7f);
    return t * logf(p) + (1.0f - t) * log1pf(-p);
}

__global__ void loss_kernel(const float* __restrict__ outg,
                            const float* __restrict__ outl,
                            const float* __restrict__ outf,
                            const float* __restrict__ target,
                            float* __restrict__ loss)
{
    __shared__ float sg[56], sl[56], sf[56];
    int t = threadIdx.x;
    if (t < 56) {
        float tg = target[t];
        sg[t] = bce_term(outg[t], tg);
        sl[t] = bce_term(outl[t], tg);
        sf[t] = bce_term(outf[t], tg);
    }
    __syncthreads();
    if (t == 0) {
        float a = 0.0f, b = 0.0f, c = 0.0f;
        for (int i = 0; i < 56; ++i) { a += sg[i]; b += sl[i]; c += sf[i]; }
        float gl = -a / 56.0f, ll = -b / 56.0f, fl = -c / 56.0f;
        loss[0] = 0.8f * gl + 0.1f * ll + 0.1f * fl;
    }
}

// ============================ host orchestration ===========================
template <int Cin, int IH, int IW, int Cout, int OH, int OW,
          int KH, int KW, int STR, int PAD>
static void launch_conv(const float* X, const float* W, float* Y, hipStream_t s)
{
    constexpr int N = OH * OW;
    dim3 grid((N + 31) / 32, Cout / 64, BATCH);
    conv_wmma_kernel<Cin, IH, IW, Cout, OH, OW, KH, KW, STR, PAD>
        <<<grid, dim3(128), 0, s>>>(X, W, Y);
}

extern "C" void kernel_launch(void* const* d_in, const int* in_sizes, int n_in,
                              void* d_out, int out_size, void* d_ws, size_t ws_size,
                              hipStream_t stream)
{
    (void)in_sizes; (void)n_in; (void)out_size; (void)ws_size;
    const float* img    = (const float*)d_in[0];
    const float* target = (const float*)d_in[1];
    const float* gw0 = (const float*)d_in[2];
    const float* gw1 = (const float*)d_in[3];
    const float* gw2 = (const float*)d_in[4];
    const float* gw3 = (const float*)d_in[5];
    const float* gfw = (const float*)d_in[6];
    const float* gfb = (const float*)d_in[7];
    const float* lw0 = (const float*)d_in[8];
    const float* lw1 = (const float*)d_in[9];
    const float* lw2 = (const float*)d_in[10];
    const float* lw3 = (const float*)d_in[11];
    const float* lfw = (const float*)d_in[12];
    const float* lfb = (const float*)d_in[13];
    const float* fw  = (const float*)d_in[14];
    const float* fb  = (const float*)d_in[15];

    float* out   = (float*)d_out;
    float* loss  = out;
    float* outg  = out + 1;
    float* outl  = out + 57;
    float* outf  = out + 113;
    float* patch = out + 169;            // [4,3,224,224]

    // workspace carve-up (4-byte units)
    float* base = (float*)d_ws;
    size_t off = 0;
    float* act0   = base + off; off += (size_t)BATCH * 64 * 56 * 56;     // 802816
    float* act1   = base + off; off += (size_t)BATCH * 512 * 28 * 28;    // 1605632
    float* act2   = base + off; off += (size_t)BATCH * 1024 * 14 * 14;   // 802816
    float* feat   = base + off; off += (size_t)BATCH * 2048 * 49;        // 401408
    float* pool_g = base + off; off += (size_t)BATCH * 2048;
    float* pool_l = base + off; off += (size_t)BATCH * 2048;
    float* hm7    = base + off; off += (size_t)BATCH * 49;
    int*   mask   = (int*)(base + off); off += (size_t)BATCH * PIX;
    int*   lab0   = (int*)(base + off); off += (size_t)BATCH * PIX;
    int*   lab1   = (int*)(base + off); off += (size_t)BATCH * PIX;
    int*   counts = (int*)(base + off); off += (size_t)BATCH * NLAB;
    int*   maxlab = (int*)(base + off); off += 4;
    int*   bbox   = (int*)(base + off); off += 16;

    // ---- global backbone ----
    launch_conv<3,    224, 224, 64,   56, 56, 7, 7, 4, 3>(img,  gw0, act0, stream);
    launch_conv<64,   56,  56,  512,  28, 28, 3, 3, 2, 1>(act0, gw1, act1, stream);
    launch_conv<512,  28,  28,  1024, 14, 14, 3, 3, 2, 1>(act1, gw2, act2, stream);
    launch_conv<1024, 14,  14,  2048, 7,  7,  3, 3, 2, 1>(act2, gw3, feat, stream);
    maxpool49_kernel<<<(BATCH * 2048 + 255) / 256, 256, 0, stream>>>(feat, pool_g, BATCH * 2048);
    fc_sigmoid_kernel<<<1, 64, 0, stream>>>(pool_g, pool_g, 2048, 0, gfw, gfb,
                                            outg, BATCH * NCLS, NCLS);

    // ---- attention patch generation ----
    hm_max_kernel<<<1, 256, 0, stream>>>(feat, hm7, BATCH * 49);
    hm_norm_kernel<<<1, 32, 0, stream>>>(hm7);
    resize_mask_kernel<<<(BATCH * PIX + 255) / 256, 256, 0, stream>>>(hm7, mask, lab0, BATCH * PIX);
    {
        int* cur = lab0; int* nxt = lab1;
        for (int it = 0; it < HW + HW; ++it) {          // 448 iterations (even)
            dilate_kernel<<<(BATCH * PIX + 255) / 256, 256, 0, stream>>>(cur, nxt, mask, BATCH * PIX);
            int* t = cur; cur = nxt; nxt = t;
        }
        // after an even number of swaps, result is back in lab0
    }
    zero_counts_kernel<<<(BATCH * NLAB + 255) / 256, 256, 0, stream>>>(counts, BATCH * NLAB);
    count_labels_kernel<<<(BATCH * PIX + 255) / 256, 256, 0, stream>>>(lab0, counts, BATCH * PIX);
    argmax_kernel<<<1, 32, 0, stream>>>(counts, maxlab, bbox);
    bbox_kernel<<<BATCH, HW, 0, stream>>>(lab0, maxlab, bbox);
    crop_resize_kernel<<<(BATCH * 3 * PIX + 255) / 256, 256, 0, stream>>>(img, bbox, patch,
                                                                          BATCH * 3 * PIX);

    // ---- local backbone (reuses activation buffers; feat_g already consumed) ----
    launch_conv<3,    224, 224, 64,   56, 56, 7, 7, 4, 3>(patch, lw0, act0, stream);
    launch_conv<64,   56,  56,  512,  28, 28, 3, 3, 2, 1>(act0,  lw1, act1, stream);
    launch_conv<512,  28,  28,  1024, 14, 14, 3, 3, 2, 1>(act1,  lw2, act2, stream);
    launch_conv<1024, 14,  14,  2048, 7,  7,  3, 3, 2, 1>(act2,  lw3, feat, stream);
    maxpool49_kernel<<<(BATCH * 2048 + 255) / 256, 256, 0, stream>>>(feat, pool_l, BATCH * 2048);
    fc_sigmoid_kernel<<<1, 64, 0, stream>>>(pool_l, pool_l, 2048, 0, lfw, lfb,
                                            outl, BATCH * NCLS, NCLS);

    // ---- fusion head + losses ----
    fc_sigmoid_kernel<<<1, 64, 0, stream>>>(pool_g, pool_l, 2048, 2048, fw, fb,
                                            outf, BATCH * NCLS, NCLS);
    loss_kernel<<<1, 64, 0, stream>>>(outg, outl, outf, target, loss);
}
